// INTP_Model_13357348290594
// MI455X (gfx1250) — matrix operand order repaired and checked by reading, not compiled, not run on hardware
//
#include <hip/hip_runtime.h>

// ---- problem dims (from reference) ----
#define B_DIM  1024
#define L_DIM  512
#define F_DIM  256
#define H_DIM  512
#define K_NN   64
#define NODES  65      // query + 64 neighbors
#define MPAD   80      // 65 padded to 5 * 16 for WMMA M tiles
#define XS     264     // padded LDS row stride (bf16 elems) to spread banks

typedef __attribute__((ext_vector_type(16))) __bf16 v16bf;
typedef __attribute__((ext_vector_type(8)))  float  v8f;

__device__ __forceinline__ unsigned short f32_to_bf16(float f) {
  unsigned u = __float_as_uint(f);
  unsigned r = 0x7fffu + ((u >> 16) & 1u);   // round-to-nearest-even
  return (unsigned short)((u + r) >> 16);
}

// ---------------------------------------------------------------------------
// Kernel 1: per-batch 64-NN selection via bitonic sort of (distbits<<32)|idx.
// Distances are >= 0 so IEEE bit pattern is order-preserving as uint.
// ---------------------------------------------------------------------------
__global__ __launch_bounds__(256)
void knn_kernel(const float* __restrict__ coords,
                const int* __restrict__ lens,
                int* __restrict__ knn) {
  __shared__ unsigned long long keys[512];
  const int b   = blockIdx.x;
  const int tid = threadIdx.x;
  const float* cb = coords + (size_t)b * L_DIM * 2;
  const float qx = cb[0], qy = cb[1];
  const int nvalid = lens[b] - 1;

  for (int c = tid; c < 512; c += 256) {
    unsigned hi = 0x7f800000u;                 // +inf sentinel (pad / masked)
    if (c < 511 && c < nvalid) {
      float dx = cb[(c + 1) * 2]     - qx;
      float dy = cb[(c + 1) * 2 + 1] - qy;
      hi = __float_as_uint(dx * dx + dy * dy);
    }
    keys[c] = ((unsigned long long)hi << 32) | (unsigned)c;
  }
  __syncthreads();

  for (unsigned k = 2; k <= 512; k <<= 1) {
    for (unsigned j = k >> 1; j > 0; j >>= 1) {
      unsigned idx = ((tid & ~(j - 1)) << 1) | (tid & (j - 1));
      unsigned ixj = idx | j;
      unsigned long long a  = keys[idx];
      unsigned long long bb = keys[ixj];
      bool up = ((idx & k) == 0);
      if ((a > bb) == up) { keys[idx] = bb; keys[ixj] = a; }
      __syncthreads();
    }
  }
  if (tid < K_NN) knn[b * K_NN + tid] = (int)(keys[tid] & 0xffffffffu);
}

// ---------------------------------------------------------------------------
// Kernel 2: W1r f32 -> bf16 (row-major [H,F]); L2-resident B operand (256 KB).
// ---------------------------------------------------------------------------
__global__ __launch_bounds__(256)
void w1r_bf16_kernel(const float* __restrict__ w, unsigned short* __restrict__ o) {
  int i = blockIdx.x * blockDim.x + threadIdx.x;
  if (i < H_DIM * F_DIM) o[i] = f32_to_bf16(w[i]);
}

// ---------------------------------------------------------------------------
// Kernel 3: per-batch SAGE forward. One workgroup (8 wave32) per batch.
// ---------------------------------------------------------------------------
__global__ __launch_bounds__(256)
void sage_kernel(const float* __restrict__ inputs,
                 const float* __restrict__ targets,
                 const float* __restrict__ W1l,
                 const float* __restrict__ b1,
                 const float* __restrict__ W2l,
                 const float* __restrict__ W2r,
                 const float* __restrict__ b2,
                 const unsigned short* __restrict__ w1rb,
                 const int* __restrict__ knn,
                 float* __restrict__ out) {
  __shared__ __align__(16) unsigned short lx[MPAD * XS];  // x as bf16, padded
  __shared__ int   lidx[NODES];
  __shared__ float lm[F_DIM];        // column mean of x
  __shared__ float lc[H_DIM];        // W1l @ m + b1
  __shared__ float la2[H_DIM];       // column sums of h (valid rows)
  __shared__ float lh0[H_DIM];       // h row 0 (query)
  __shared__ float lred[256];

  const int b    = blockIdx.x;
  const int tid  = threadIdx.x;
  const int lane = tid & 31;
  const int wave = tid >> 5;

  if (tid < K_NN) lidx[tid + 1] = knn[b * K_NN + tid];
  if (tid == K_NN) lidx[0] = 0;
  la2[tid] = 0.0f;
  la2[tid + 256] = 0.0f;
  __builtin_prefetch(w1rb + (size_t)tid * 512, 0, 1);  // warm B operand (global_prefetch_b8)
  __syncthreads();

  // ---- gather 65 rows, convert f32 -> bf16 into LDS (rows >= 65 zeroed) ----
  const float* xin = inputs + (size_t)b * L_DIM * F_DIM;
  for (int r0 = 0; r0 < MPAD; r0 += 4) {
    int r  = r0 + (tid >> 6);
    int cg = (tid & 63) * 4;
    float4 v = make_float4(0.f, 0.f, 0.f, 0.f);
    if (r < NODES) {
      int src = lidx[r];
      v = *(const float4*)(xin + (size_t)src * F_DIM + cg);
    }
    ushort4 pk;
    pk.x = f32_to_bf16(v.x); pk.y = f32_to_bf16(v.y);
    pk.z = f32_to_bf16(v.z); pk.w = f32_to_bf16(v.w);
    *(ushort4*)&lx[r * XS + cg] = pk;
  }
  __syncthreads();

  // ---- column mean over the 65 valid rows (thread t owns column t) ----
  {
    float s = 0.f;
    for (int r = 0; r < NODES; ++r) {
      unsigned short h = lx[r * XS + tid];
      s += __uint_as_float((unsigned)h << 16);
    }
    lm[tid] = s * (1.0f / NODES);
  }
  __syncthreads();

  // ---- c[n] = W1l[n,:] . m + b1[n]  (W1l L2-resident, 512 KB) ----
  for (int n = tid; n < H_DIM; n += 256) {
    const float4* wr = (const float4*)(W1l + (size_t)n * F_DIM);
    float s = b1[n];
#pragma unroll 4
    for (int f4 = 0; f4 < F_DIM / 4; ++f4) {
      float4 w = wr[f4];
      s += w.x * lm[f4 * 4] + w.y * lm[f4 * 4 + 1] +
           w.z * lm[f4 * 4 + 2] + w.w * lm[f4 * 4 + 3];
    }
    lc[n] = s;
  }
  __syncthreads();

  // ---- main GEMM: h = relu(x @ W1r^T + c) ----
  // 5 M-tiles x 32 N-tiles of 16x16, K = 256 (8 WMMA steps).
  // Each wave owns 4 N-tiles (wave, wave+8, wave+16, wave+24) per M-tile and
  // keeps 4 accumulators live; the A-fragment is loaded once per K-step and
  // shared across the 4 WMMAs (low register pressure, 4x A reuse).
  const int half = lane >> 4;   // 0: lanes 0-15, 1: lanes 16-31
  const int l16  = lane & 15;
  union AU { uint4 u[2]; v16bf v; };
  // distance between the wave's consecutive B-fragment loads: 4 tiles * 16 rows * F
  // = 16384 bf16 elements = 2048 uint4 = 32 KB (fits the 24-bit ioffset).
  const unsigned short* bbase = w1rb + (size_t)(wave * 16 + l16) * F_DIM;

#pragma unroll 1
  for (int mT = 0; mT < 5; ++mT) {
    const int arow = mT * 16 + l16;   // A: lane%16 selects M row

    v8f acc[4] = {{}, {}, {}, {}};
#pragma unroll
    for (int kt = 0; kt < 8; ++kt) {
      // A 16x32 bf16: lanes 0-15 hold K 0-7 & 16-23, lanes 16-31 K 8-15 & 24-31
      AU A;
      const int ka = kt * 32 + half * 8;
      A.u[0] = *(const uint4*)&lx[arow * XS + ka];
      A.u[1] = *(const uint4*)&lx[arow * XS + ka + 16];

      // B 32x16 bf16: lanes 0-15 hold K 0-7 & 16-23, lanes 16-31 K 8-15 & 24-31
      const int kb = kt * 32 + half * 8;
      const uint4* bp = (const uint4*)(bbase + kb);
      AU Bf[4];
#pragma unroll
      for (int nn = 0; nn < 4; ++nn) {
        Bf[nn].u[0] = bp[nn * 2048];
        Bf[nn].u[1] = bp[nn * 2048 + 2];
      }
#pragma unroll
      for (int nn = 0; nn < 4; ++nn) {
        acc[nn] = __builtin_amdgcn_wmma_f32_16x16x32_bf16(
            false, A.v, false, Bf[nn].v, (short)0, acc[nn], false, false);
      }
    }

    // epilogue: +c, relu, capture row 0, accumulate valid-row column sums.
    // Rows of mT 0..3 are all < 65; for mT == 4 only row 64 (v==0, half==0).
#pragma unroll
    for (int nn = 0; nn < 4; ++nn) {
      const int nT = wave + nn * 8;             // 0..31
      const int nc = nT * 16 + l16;             // this lane's column
      const float cn = lc[nc];
      const float h0 = fmaxf(acc[nn][0] + cn, 0.f);
      float colsum = 0.f;
      if (mT < 4) {
        colsum = h0;
#pragma unroll
        for (int v = 1; v < 8; ++v) colsum += fmaxf(acc[nn][v] + cn, 0.f);
      } else if (half == 0) {
        colsum = h0;                            // row 64 only
      }
      if (mT == 0 && half == 0) lh0[nc] = h0;   // query row
      atomicAdd(&la2[nc], colsum);              // ds_add_f32
    }
  }
  __syncthreads();

  // ---- out0 = W2l . mean(h) + W2r . h0 + b2 ; y0 = targets[b,0] ----
  float p = 0.f;
  for (int n = tid; n < H_DIM; n += 256)
    p += W2l[n] * (la2[n] * (1.0f / NODES)) + W2r[n] * lh0[n];
  lred[tid] = p;
  __syncthreads();
  for (int s = 128; s > 0; s >>= 1) {
    if (tid < s) lred[tid] += lred[tid + s];
    __syncthreads();
  }
  if (tid == 0) {
    out[b]         = lred[0] + b2[0];
    out[B_DIM + b] = targets[(size_t)b * L_DIM];
  }
}

// ---------------------------------------------------------------------------
extern "C" void kernel_launch(void* const* d_in, const int* in_sizes, int n_in,
                              void* d_out, int out_size, void* d_ws, size_t ws_size,
                              hipStream_t stream) {
  const float* inputs  = (const float*)d_in[0];
  const float* coords  = (const float*)d_in[1];
  const float* targets = (const float*)d_in[2];
  const int*   lens    = (const int*)d_in[3];
  const float* W1l     = (const float*)d_in[4];
  const float* W1r     = (const float*)d_in[5];
  const float* b1      = (const float*)d_in[6];
  const float* W2l     = (const float*)d_in[7];
  const float* W2r     = (const float*)d_in[8];
  const float* b2      = (const float*)d_in[9];
  float* out = (float*)d_out;

  // workspace: [1024*64 int knn idx][512*256 bf16 W1r]  = 512 KB total
  int* knn = (int*)d_ws;
  unsigned short* w1rb =
      (unsigned short*)((char*)d_ws + (size_t)B_DIM * K_NN * sizeof(int));

  knn_kernel<<<B_DIM, 256, 0, stream>>>(coords, lens, knn);
  w1r_bf16_kernel<<<(H_DIM * F_DIM + 255) / 256, 256, 0, stream>>>(W1r, w1rb);
  sage_kernel<<<B_DIM, 256, 0, stream>>>(inputs, targets, W1l, b1, W2l, W2r, b2,
                                         w1rb, knn, out);
}
